// FluteLinear_75462575390751
// MI455X (gfx1250) — matrix operand order repaired and checked by reading, not compiled, not run on hardware
//
#include <hip/hip_runtime.h>

#define M_DIM 4096
#define N_DIM 4096
#define K_DIM 4096
#define GROUP 128
#define G_DIM (K_DIM / GROUP)   // 32

typedef __attribute__((ext_vector_type(16))) __bf16 v16bf;
typedef __attribute__((ext_vector_type(8)))  float  v8f;

union FragBF { uint4 u2[2]; v16bf v; };

// Pack two f32 -> two bf16 in one dword; native __bf16 casts let the backend
// use hardware cvt (v_cvt_pk_bf16_f32 class) instead of manual rounding VALU.
__device__ __forceinline__ unsigned int pk2(float lo, float hi) {
  union { __bf16 b[2]; unsigned int u; } r;
  r.b[0] = (__bf16)lo; r.b[1] = (__bf16)hi;
  return r.u;
}
__device__ __forceinline__ uint2 cv4(float4 a) {
  uint2 r; r.x = pk2(a.x, a.y); r.y = pk2(a.z, a.w); return r;
}
__device__ __forceinline__ uint2 dq4(int4 q, float s, const float* tab) {
  uint2 r;
  r.x = pk2(tab[q.x & 15] * s, tab[q.y & 15] * s);
  r.y = pk2(tab[q.z & 15] * s, tab[q.w & 15] * s);
  return r;
}

// ---------------------------------------------------------------------------
// Kernel 1: A [M,K] fp32 -> bf16 (memory bound)
// ---------------------------------------------------------------------------
__global__ __launch_bounds__(256)
void cvt_a_bf16(const float* __restrict__ x, unsigned short* __restrict__ a16) {
  const size_t i = ((size_t)blockIdx.x * 256 + threadIdx.x) * 8;
  float4 f0 = *(const float4*)(x + i);
  float4 f1 = *(const float4*)(x + i + 4);
  uint2 p0 = cv4(f0), p1 = cv4(f1);
  *(uint4*)(a16 + i) = make_uint4(p0.x, p0.y, p1.x, p1.y);
}

// ---------------------------------------------------------------------------
// Kernel 2: W dequant: w16[n,k] = bf16(tables[q[n,k]] * scales[n,k/128])
// ---------------------------------------------------------------------------
__global__ __launch_bounds__(256)
void dq_w_bf16(const int* __restrict__ qw, const float* __restrict__ scales,
               const float* __restrict__ tables, unsigned short* __restrict__ w16) {
  __shared__ float tab[16];
  if (threadIdx.x < 16) tab[threadIdx.x] = tables[threadIdx.x];
  __syncthreads();
  const size_t i = ((size_t)blockIdx.x * 256 + threadIdx.x) * 16;
  const int n = (int)(i >> 12);       // / K_DIM
  const int k = (int)(i & 4095);      // % K_DIM  (16 elems stay in one group)
  const float s = scales[(size_t)n * G_DIM + (k >> 7)];
  const int4* qp = (const int4*)(qw + i);
  int4 q0 = qp[0], q1 = qp[1], q2 = qp[2], q3 = qp[3];
  uint2 b0 = dq4(q0, s, tab), b1 = dq4(q1, s, tab);
  uint2 b2 = dq4(q2, s, tab), b3 = dq4(q3, s, tab);
  *(uint4*)(w16 + i)     = make_uint4(b0.x, b0.y, b1.x, b1.y);
  *(uint4*)(w16 + i + 8) = make_uint4(b2.x, b2.y, b3.x, b3.y);
}

// ---------------------------------------------------------------------------
// Kernel 3: pure bf16 WMMA GEMM, async global->LDS copies (ASYNCcnt path).
// Block 256 thr / 8 waves, C tile 128x128, BK=64, double-buffered LDS.
// ---------------------------------------------------------------------------
#define BK3 64
#define NKT3 (K_DIM / BK3)   // 64

// Copy 64 contiguous bytes global->LDS per lane. Per ISA §10: the instruction
// offset is added to BOTH the global address and the LDS address, so one
// base pair + imm offsets walks both sides together.
#define ASYNC_CP64(ldsoff, gaddr)                                           \
  asm volatile("global_load_async_to_lds_b128 %0, %1, off\n\t"              \
               "global_load_async_to_lds_b128 %0, %1, off offset:16\n\t"    \
               "global_load_async_to_lds_b128 %0, %1, off offset:32\n\t"    \
               "global_load_async_to_lds_b128 %0, %1, off offset:48"        \
               :: "v"(ldsoff), "v"(gaddr) : "memory")

__global__ __launch_bounds__(256)
void gemm_bf16_wmma_async(const unsigned short* __restrict__ A16,
                          const unsigned short* __restrict__ W16,
                          float* __restrict__ out) {
  __shared__ unsigned short As[2][128 * BK3];   // 16 KB per buffer
  __shared__ unsigned short Bs[2][128 * BK3];   // 16 KB per buffer

  const int tid  = threadIdx.x;
  const int lane = tid & 31;
  const int wave = tid >> 5;
  const int bm   = blockIdx.y * 128;
  const int bn   = blockIdx.x * 128;

  // copy mapping: 2 threads per tile row, 32 bf16 (64 B) each
  const int crow = tid >> 1;
  const int ccol = (tid & 1) * 32;
  const unsigned short* aG = A16 + (size_t)(bm + crow) * K_DIM + ccol;
  const unsigned short* bG = W16 + (size_t)(bn + crow) * K_DIM + ccol;
  const unsigned aL0 = (unsigned)(uintptr_t)&As[0][crow * BK3 + ccol];
  const unsigned aL1 = (unsigned)(uintptr_t)&As[1][crow * BK3 + ccol];
  const unsigned bL0 = (unsigned)(uintptr_t)&Bs[0][crow * BK3 + ccol];
  const unsigned bL1 = (unsigned)(uintptr_t)&Bs[1][crow * BK3 + ccol];

  // wave tile: 64(M) x 32(N) -> 4x2 fragments
  const int wm   = (wave >> 2) * 64;
  const int wn   = (wave & 3) * 32;
  const int rA   = lane & 15;
  const int kb8  = (lane >> 4) * 8;
  const int kb16 = (lane >> 4) * 16;

  v8f acc[4][2] = {};

  // prologue: async-copy tile 0 into buffer 0
  ASYNC_CP64(aL0, (unsigned long long)(uintptr_t)aG);
  ASYNC_CP64(bL0, (unsigned long long)(uintptr_t)bG);

  int buf = 0;
  for (int kt = 0; kt < NKT3; ++kt) {
    asm volatile("s_wait_asynccnt 0" ::: "memory");  // own copies landed
    __syncthreads();                                 // all waves' copies visible

    if (kt + 1 < NKT3) {                             // overlap next tile's DMA
      const unsigned long long ga =
          (unsigned long long)(uintptr_t)(aG + (size_t)(kt + 1) * BK3);
      const unsigned long long gb =
          (unsigned long long)(uintptr_t)(bG + (size_t)(kt + 1) * BK3);
      ASYNC_CP64(buf ? aL0 : aL1, ga);
      ASYNC_CP64(buf ? bL0 : bL1, gb);
    }

    #pragma unroll
    for (int ks = 0; ks < 2; ++ks) {                 // two K=32 steps per tile
      FragBF afr[4], bfr[2];
      #pragma unroll
      for (int i = 0; i < 4; ++i) {
        const unsigned short* p = &As[buf][(wm + i * 16 + rA) * BK3 + ks * 32 + kb8];
        afr[i].u2[0] = *(const uint4*)p;
        afr[i].u2[1] = *(const uint4*)(p + 16);
      }
      #pragma unroll
      for (int j = 0; j < 2; ++j) {
        const unsigned short* p = &Bs[buf][(wn + j * 16 + rA) * BK3 + ks * 32 + kb8];
        bfr[j].u2[0] = *(const uint4*)p;
        bfr[j].u2[1] = *(const uint4*)(p + 16);
      }
      #pragma unroll
      for (int i = 0; i < 4; ++i)
        #pragma unroll
        for (int j = 0; j < 2; ++j)
          acc[i][j] = __builtin_amdgcn_wmma_f32_16x16x32_bf16(
              false, afr[i].v, false, bfr[j].v, (short)0, acc[i][j], false, false);
    }
    buf ^= 1;
  }

  float* oBase = out + (size_t)(bm + wm) * N_DIM + bn + wn;
  const int mOff = (lane >> 4) * 8;
  const int nOff = lane & 15;
  #pragma unroll
  for (int i = 0; i < 4; ++i)
    #pragma unroll
    for (int j = 0; j < 2; ++j)
      #pragma unroll
      for (int r = 0; r < 8; ++r)
        oBase[(size_t)(i * 16 + mOff + r) * N_DIM + j * 16 + nOff] = acc[i][j][r];
}

// ---------------------------------------------------------------------------
// Fallback: fused dequant+GEMM (used when d_ws is too small for bf16 staging)
// ---------------------------------------------------------------------------
#define BM 128
#define BN 128
#define BK 32
#define NKT (K_DIM / BK)

__global__ __launch_bounds__(256, 1)
void flute_qgemm_bf16_fused(const float* __restrict__ x,
                            const int*   __restrict__ qw,
                            const float* __restrict__ scales,
                            const float* __restrict__ tables,
                            float* __restrict__ out) {
  __shared__ unsigned short As[2][BM * BK];
  __shared__ unsigned short Bs[2][BN * BK];
  __shared__ float tab[16];

  const int tid  = threadIdx.x;
  const int lane = tid & 31;
  const int wave = tid >> 5;
  const int bm   = blockIdx.y * BM;
  const int bn   = blockIdx.x * BN;

  if (tid < 16) tab[tid] = tables[tid];

  const int lrow = tid >> 1;
  const int lk   = (tid & 1) * 16;
  const float* aBase = x  + (size_t)(bm + lrow) * K_DIM + lk;
  const int*   qBase = qw + (size_t)(bn + lrow) * K_DIM + lk;
  const float* sBase = scales + (size_t)(bn + lrow) * G_DIM;

  const int wm   = (wave >> 2) * 64;
  const int wn   = (wave & 3) * 32;
  const int rA   = lane & 15;
  const int kb8  = (lane >> 4) * 8;
  const int kb16 = (lane >> 4) * 16;

  v8f acc[4][2] = {};
  float4 ar[4];
  int4   qr[4];

  #pragma unroll
  for (int v = 0; v < 4; ++v) ar[v] = ((const float4*)aBase)[v];
  #pragma unroll
  for (int v = 0; v < 4; ++v) qr[v] = ((const int4*)qBase)[v];

  __syncthreads();

  {
    const float s = sBase[0];
    uint2 a0 = cv4(ar[0]), a1 = cv4(ar[1]), a2 = cv4(ar[2]), a3 = cv4(ar[3]);
    *(uint4*)&As[0][lrow * BK + lk]     = make_uint4(a0.x, a0.y, a1.x, a1.y);
    *(uint4*)&As[0][lrow * BK + lk + 8] = make_uint4(a2.x, a2.y, a3.x, a3.y);
    uint2 b0 = dq4(qr[0], s, tab), b1 = dq4(qr[1], s, tab);
    uint2 b2 = dq4(qr[2], s, tab), b3 = dq4(qr[3], s, tab);
    *(uint4*)&Bs[0][lrow * BK + lk]     = make_uint4(b0.x, b0.y, b1.x, b1.y);
    *(uint4*)&Bs[0][lrow * BK + lk + 8] = make_uint4(b2.x, b2.y, b3.x, b3.y);
  }

  int buf = 0;
  for (int kt = 0; kt < NKT; ++kt) {
    __syncthreads();

    const bool more = (kt + 1) < NKT;
    if (more) {
      const float* ap = aBase + (size_t)(kt + 1) * BK;
      const int*   qp = qBase + (size_t)(kt + 1) * BK;
      #pragma unroll
      for (int v = 0; v < 4; ++v) ar[v] = ((const float4*)ap)[v];
      #pragma unroll
      for (int v = 0; v < 4; ++v) qr[v] = ((const int4*)qp)[v];
    }

    FragBF afr[4], bfr[2];
    #pragma unroll
    for (int i = 0; i < 4; ++i) {
      const unsigned short* p = &As[buf][(wm + i * 16 + rA) * BK + kb8];
      afr[i].u2[0] = *(const uint4*)p;
      afr[i].u2[1] = *(const uint4*)(p + 16);
    }
    #pragma unroll
    for (int j = 0; j < 2; ++j) {
      const unsigned short* p = &Bs[buf][(wn + j * 16 + rA) * BK + kb8];
      bfr[j].u2[0] = *(const uint4*)p;
      bfr[j].u2[1] = *(const uint4*)(p + 16);
    }
    #pragma unroll
    for (int i = 0; i < 4; ++i)
      #pragma unroll
      for (int j = 0; j < 2; ++j)
        acc[i][j] = __builtin_amdgcn_wmma_f32_16x16x32_bf16(
            false, afr[i].v, false, bfr[j].v, (short)0, acc[i][j], false, false);

    if (more) {
      const float s = sBase[(kt + 1) >> 2];
      uint2 a0 = cv4(ar[0]), a1 = cv4(ar[1]), a2 = cv4(ar[2]), a3 = cv4(ar[3]);
      *(uint4*)&As[buf ^ 1][lrow * BK + lk]     = make_uint4(a0.x, a0.y, a1.x, a1.y);
      *(uint4*)&As[buf ^ 1][lrow * BK + lk + 8] = make_uint4(a2.x, a2.y, a3.x, a3.y);
      uint2 b0 = dq4(qr[0], s, tab), b1 = dq4(qr[1], s, tab);
      uint2 b2 = dq4(qr[2], s, tab), b3 = dq4(qr[3], s, tab);
      *(uint4*)&Bs[buf ^ 1][lrow * BK + lk]     = make_uint4(b0.x, b0.y, b1.x, b1.y);
      *(uint4*)&Bs[buf ^ 1][lrow * BK + lk + 8] = make_uint4(b2.x, b2.y, b3.x, b3.y);
    }
    buf ^= 1;
  }

  float* oBase = out + (size_t)(bm + wm) * N_DIM + bn + wn;
  const int mOff = (lane >> 4) * 8;
  const int nOff = lane & 15;
  #pragma unroll
  for (int i = 0; i < 4; ++i)
    #pragma unroll
    for (int j = 0; j < 2; ++j)
      #pragma unroll
      for (int r = 0; r < 8; ++r)
        oBase[(size_t)(i * 16 + mOff + r) * N_DIM + j * 16 + nOff] = acc[i][j][r];
}

// ---------------------------------------------------------------------------
extern "C" void kernel_launch(void* const* d_in, const int* in_sizes, int n_in,
                              void* d_out, int out_size, void* d_ws, size_t ws_size,
                              hipStream_t stream) {
  const float* x   = (const float*)d_in[0];   // [M,K] fp32
  const int*   qw  = (const int*)  d_in[1];   // [N,K] int32 (4-bit codes)
  const float* sc  = (const float*)d_in[2];   // [N,G] fp32
  const float* tb  = (const float*)d_in[3];   // [16] fp32
  float*       out = (float*)d_out;           // [M,N] fp32

  const size_t needA = (size_t)M_DIM * K_DIM * sizeof(unsigned short);  // 32 MB
  const size_t needW = (size_t)N_DIM * K_DIM * sizeof(unsigned short);  // 32 MB

  if (ws_size >= needA + needW) {
    unsigned short* a16 = (unsigned short*)d_ws;
    unsigned short* w16 = (unsigned short*)((char*)d_ws + needA);
    cvt_a_bf16<<<(M_DIM * (size_t)K_DIM) / (256 * 8), 256, 0, stream>>>(x, a16);
    dq_w_bf16 <<<(N_DIM * (size_t)K_DIM) / (256 * 16), 256, 0, stream>>>(qw, sc, tb, w16);
    dim3 grid(N_DIM / 128, M_DIM / 128);
    gemm_bf16_wmma_async<<<grid, 256, 0, stream>>>(a16, w16, out);
  } else {
    dim3 grid(N_DIM / BN, M_DIM / BM);
    flute_qgemm_bf16_fused<<<grid, 256, 0, stream>>>(x, qw, sc, tb, out);
  }

  (void)in_sizes; (void)n_in; (void)out_size;
}
